// LatticeLSTM_SeqLabel_51711406244465
// MI455X (gfx1250) — matrix operand order repaired and checked by reading, not compiled, not run on hardware
//
#include <hip/hip_runtime.h>
#include <hip/hip_bf16.h>
#include <math.h>

// ---------------- problem constants ----------------
#define BN   16
#define TN   128
#define LN   10
#define HN   300
#define H3   900
#define NTAG 18

#define KP_H 320   // H=300 padded to x32
#define KP_E 128   // (Dc+Db)=100 padded
#define KP_W 64    // Dw=50 padded
// N paddings to multiples of 64 so 4-tile strips never load OOB B rows
#define NP_H3 960
#define NS_H3 15   // 960/64 n-strips
#define NP_H 320
#define NS_H 5
#define NP_TAG 64
#define NS_TAG 1

typedef __attribute__((ext_vector_type(16))) _Float16 v16h;
typedef __attribute__((ext_vector_type(8)))  _Float16 v8h;
typedef __attribute__((ext_vector_type(8)))  float    v8f;

__device__ __forceinline__ float sigmoidf_(float x) { return 1.0f / (1.0f + expf(-x)); }

// ---------------- WMMA strip GEMM: MB x 4 tiles of 16x16 per wave ----------------
// C = A(f16,[M,Kp]) x Bt(f16,[Npad,Kp])^T.  A rows mult of 16, Kp mult of 32, Npad mult of 64.
// A fragment reused across 4 n-tiles, B fragment reused across MB m-tiles.
template <int MB>
__device__ __forceinline__ void gemm_strip(const _Float16* __restrict__ A, int lda,
                                           const _Float16* __restrict__ Bt, int ldb,
                                           const float* __restrict__ bias,
                                           float* Cf, _Float16* Ch, int ldc,
                                           int M, int N, int Kp, int ms, int ns, int lane) {
  const int r    = lane & 15;
  const int half = lane >> 4;
  const _Float16* Ar[MB];
#pragma unroll
  for (int i = 0; i < MB; ++i) Ar[i] = A + (size_t)(ms * MB * 16 + i * 16 + r) * lda;
  const _Float16* Br = Bt + (size_t)(ns * 64 + r) * ldb;
  v8f acc[MB][4] = {};
  for (int k = 0; k < Kp; k += 32) {
    v16h a[MB];
#pragma unroll
    for (int i = 0; i < MB; ++i) {
      // A fragment: lanes0-15: K {k..k+15}; lanes16-31: {k+16..k+31}
      v8h lo = *(const v8h*)(Ar[i] + k + half * 16);
      v8h hi = *(const v8h*)(Ar[i] + k + half * 16 + 8);
      a[i] = __builtin_shufflevector(lo, hi, 0,1,2,3,4,5,6,7,8,9,10,11,12,13,14,15);
    }
#pragma unroll
    for (int j = 0; j < 4; ++j) {
      // B fragment: lane = N column (Bt row), half selects K 0-15 vs 16-31
      v16h b = *(const v16h*)(Br + (size_t)j * 16 * ldb + k + half * 16);
#pragma unroll
      for (int i = 0; i < MB; ++i)
        acc[i][j] = __builtin_amdgcn_wmma_f32_16x16x32_f16(false, a[i], false, b,
                                                           (short)0, acc[i][j], false, false);
    }
  }
#pragma unroll
  for (int j = 0; j < 4; ++j) {
    const int nn = ns * 64 + j * 16 + r;
    if (nn < N) {
      const float bv = bias ? bias[nn] : 0.0f;
#pragma unroll
      for (int i = 0; i < MB; ++i) {
        const int mb_ = ms * MB * 16 + i * 16 + half * 8;
#pragma unroll
        for (int q = 0; q < 8; ++q) {
          const int mm = mb_ + q;
          if (mm < M) {
            const float v = acc[i][j][q] + bv;
            if (Cf) Cf[(size_t)mm * ldc + nn] = v;
            else    Ch[(size_t)mm * ldc + nn] = (_Float16)v;
          }
        }
      }
    }
  }
}

__global__ __launch_bounds__(256) void wmma_gemm_kernel(const _Float16* __restrict__ A, int lda,
                                                        const _Float16* __restrict__ Bt, int ldb,
                                                        const float* __restrict__ bias,
                                                        float* Cf, _Float16* Ch, int ldc,
                                                        int M, int N, int Kp, int mstrips, int nstrips) {
  const int lane = threadIdx.x & 31;
  const int gw = blockIdx.x * (blockDim.x >> 5) + (threadIdx.x >> 5);
  if (gw >= mstrips * nstrips) return;
  gemm_strip<1>(A, lda, Bt, ldb, bias, Cf, Ch, ldc, M, N, Kp, gw / nstrips, gw % nstrips, lane);
}

// ---------------- prep kernels ----------------
__global__ void zero_u32_kernel(unsigned int* p, size_t n) {
  size_t i = (size_t)blockIdx.x * blockDim.x + threadIdx.x;
  if (i < n) p[i] = 0u;
}

// W [K,N] row-major f32  ->  Bt [Npad, Kpad] row-major f16 (zero padded)
__global__ void transpose_f16_kernel(const float* __restrict__ W, _Float16* __restrict__ Bt,
                                     int K, int N, int Kpad, int Npad) {
  int i = blockIdx.x * blockDim.x + threadIdx.x;
  if (i >= Npad * Kpad) return;
  int n = i / Kpad, k = i % Kpad;
  Bt[i] = (n < N && k < K) ? (_Float16)W[(size_t)k * N + n] : (_Float16)0.0f;
}

// embA [B*T, 128] = concat(char_emb[50], bigram_emb[50], pad)
__global__ void gather_emb_kernel(const int* __restrict__ chars, const int* __restrict__ bigrams,
                                  const float* __restrict__ ct, const float* __restrict__ bt,
                                  _Float16* __restrict__ embA) {
  int i = blockIdx.x * blockDim.x + threadIdx.x;
  if (i >= BN * TN * KP_E) return;
  int row = i >> 7, k = i & 127;
  float v = 0.0f;
  if (k < 50)        v = ct[(size_t)chars[row] * 50 + k];
  else if (k < 100)  v = bt[(size_t)bigrams[row] * 50 + (k - 50)];
  embA[i] = (_Float16)v;
}

// wembA [B*T*L, 64] = word_emb[50] + pad
__global__ void gather_wemb_kernel(const int* __restrict__ words, const float* __restrict__ wt,
                                   _Float16* __restrict__ wembA) {
  int i = blockIdx.x * blockDim.x + threadIdx.x;
  if (i >= BN * TN * LN * KP_W) return;
  int row = i >> 6, k = i & 63;
  wembA[i] = (k < 50) ? (_Float16)wt[(size_t)words[row] * 50 + k] : (_Float16)0.0f;
}

// ---------------- sequential lattice-LSTM scan: 1 persistent workgroup, 16 waves ----------------
__global__ __launch_bounds__(512, 1) void lattice_scan_kernel(
    const int* __restrict__ skips_src, const int* __restrict__ lex_cnt,
    const _Float16* __restrict__ wT_wwhh, const _Float16* __restrict__ wT_awhh,
    const _Float16* __restrict__ wT_whh,  const _Float16* __restrict__ wwi_h,
    const float* __restrict__ wi_all, const float* __restrict__ awi_all,
    float* __restrict__ Hbuf, float* __restrict__ Cbuf, _Float16* __restrict__ hsA,
    _Float16* __restrict__ hbA, _Float16* __restrict__ cwA, _Float16* __restrict__ hA,
    float* __restrict__ cb, float* __restrict__ wordsum, float* __restrict__ cw,
    float* __restrict__ alphaRaw, float* __restrict__ charsum) {
  const int tid = threadIdx.x;
  const int lane = tid & 31;
  const int wid = tid >> 5;
  const int NT = 512, NW = 16;

  for (int t = 0; t < TN; ++t) {
    // Phase A: gather lattice predecessor states into f16 A-matrices
    for (int i = tid; i < 160 * KP_H; i += NT) {
      int r = i / KP_H, k = i - r * KP_H;
      int b = r / LN, l = r - b * LN;
      int src = skips_src[((size_t)b * TN + t) * LN + l];
      int idx = src + 1; idx = idx < 0 ? 0 : (idx > TN ? TN : idx);
      float hv = 0.0f;
      if (k < HN) {
        size_t o = ((size_t)b * (TN + 1) + idx) * HN + k;
        hv = Hbuf[o];
        cb[r * HN + k] = Cbuf[o];
      }
      hbA[i] = (_Float16)hv;
    }
    for (int i = tid; i < BN * KP_H; i += NT) {
      int b = i / KP_H, k = i - b * KP_H;
      float hv = (k < HN) ? Hbuf[((size_t)b * (TN + 1) + t) * HN + k] : 0.0f;
      hA[i] = (_Float16)hv;
    }
    // Prefetch step t+1 gate inputs into cache while this step computes
    if (t + 1 < TN) {
      for (int idx = tid * 64; idx < 160 * H3; idx += NT * 64) {       // wwi: 128B lines (f16)
        int r = idx / H3, e = idx - r * H3;
        int b = r / LN, l = r - b * LN;
        __builtin_prefetch(wwi_h + ((size_t)(b * TN + t + 1) * LN + l) * H3 + e, 0, 1);
      }
      for (int idx = tid * 32; idx < BN * H3; idx += NT * 32) {        // wi: 128B lines (f32)
        int b = idx / H3, e = idx - b * H3;
        __builtin_prefetch(wi_all + (size_t)(b * TN + t + 1) * H3 + e, 0, 1);
      }
      for (int idx = tid * 32; idx < BN * HN; idx += NT * 32) {        // awi
        int b = idx / HN, e = idx - b * HN;
        __builtin_prefetch(awi_all + (size_t)(b * TN + t + 1) * HN + e, 0, 1);
      }
    }
    __syncthreads();
    // Phase B: wordsum[160,900] = hbA x word_w_hh^T  (5 mstrips x 15 nstrips, 2x4 tiles each)
    for (int s = wid; s < 5 * NS_H3; s += NW)
      gemm_strip<2>(hbA, KP_H, wT_wwhh, KP_H, nullptr, wordsum, nullptr, H3,
                    160, H3, KP_H, s / NS_H3, s % NS_H3, lane);
    __syncthreads();
    // Phase C: word cell: cw = sig(f)*cb + sig(i)*tanh(g)
    for (int i = tid; i < 160 * KP_H; i += NT) {
      int r = i / KP_H, k = i - r * KP_H;
      float cwv = 0.0f;
      if (k < HN) {
        int b = r / LN, l = r - b * LN;
        size_t wrow = ((size_t)(b * TN + t) * LN + l) * H3;
        float f  = sigmoidf_((float)wwi_h[wrow + k]          + wordsum[r * H3 + k]);
        float iw = sigmoidf_((float)wwi_h[wrow + HN + k]     + wordsum[r * H3 + HN + k]);
        float g  = tanhf(    (float)wwi_h[wrow + 2 * HN + k] + wordsum[r * H3 + 2 * HN + k]);
        cwv = f * cb[r * HN + k] + iw * g;
        cw[r * HN + k] = cwv;
      }
      cwA[i] = (_Float16)cwv;
    }
    __syncthreads();
    // Phase D: alphaRaw[160,300] = cwA x alpha_w_hh^T ; Phase E: charsum[16,900] = hA x w_hh^T
    for (int s = wid; s < 5 * NS_H; s += NW)
      gemm_strip<2>(cwA, KP_H, wT_awhh, KP_H, nullptr, alphaRaw, nullptr, HN,
                    160, HN, KP_H, s / NS_H, s % NS_H, lane);
    for (int s = wid; s < NS_H3; s += NW)
      gemm_strip<1>(hA, KP_H, wT_whh, KP_H, nullptr, charsum, nullptr, H3,
                    BN, H3, KP_H, 0, s, lane);
    __syncthreads();
    // Phase F: gate merge + state update
    for (int i = tid; i < BN * HN; i += NT) {
      int b = i / HN, k = i - b * HN;
      size_t wrow = (size_t)(b * TN + t) * H3;
      float ig = sigmoidf_(wi_all[wrow + k]          + charsum[b * H3 + k]);
      float og = sigmoidf_(wi_all[wrow + HN + k]     + charsum[b * H3 + HN + k]);
      float gg = tanhf(    wi_all[wrow + 2 * HN + k] + charsum[b * H3 + 2 * HN + k]);
      int cnt = lex_cnt[b * TN + t];
      float c_new;
      if (cnt == 0) {
        float cp = Cbuf[((size_t)b * (TN + 1) + t) * HN + k];
        c_new = (1.0f - ig) * cp + ig * gg;
      } else {
        float e = expf(ig);
        float wsum = e, acc = e * gg;
        float aw = awi_all[(size_t)(b * TN + t) * HN + k];
        for (int l = 0; l < cnt; ++l) {
          int r = b * LN + l;
          float ee = expf(sigmoidf_(aw + alphaRaw[r * HN + k]));
          wsum += ee; acc += ee * cw[r * HN + k];
        }
        c_new = acc / wsum;
      }
      float h_new = og * tanhf(c_new);
      size_t o = ((size_t)b * (TN + 1) + t + 1) * HN + k;
      Hbuf[o] = h_new; Cbuf[o] = c_new;
      hsA[(size_t)(b * TN + t) * KP_H + k] = (_Float16)h_new;
    }
    __syncthreads();
  }
}

// ---------------- CRF forward + gold score ----------------
__global__ __launch_bounds__(512) void crf_kernel(const float* __restrict__ feats,
                                                  const float* __restrict__ trans,
                                                  const float* __restrict__ startS,
                                                  const float* __restrict__ endS,
                                                  const int* __restrict__ seq_len,
                                                  const int* __restrict__ target,
                                                  float* __restrict__ out) {
  __shared__ float sa[BN][NTAG];
  const int tid = threadIdx.x;
  const int b = tid >> 5, j = tid & 31;
  const bool active = (b < BN) && (j < NTAG);
  if (active) sa[b][j] = feats[(size_t)(b * TN) * NTAG + j] + startS[j];
  __syncthreads();
  for (int t = 1; t < TN; ++t) {
    float val = 0.0f;
    bool mt = false;
    if (active) {
      mt = t < seq_len[b];
      float mx = -1e30f;
      for (int i = 0; i < NTAG; ++i) mx = fmaxf(mx, sa[b][i] + trans[i * NTAG + j]);
      float s = 0.0f;
      for (int i = 0; i < NTAG; ++i) s += expf(sa[b][i] + trans[i * NTAG + j] - mx);
      val = mx + logf(s) + feats[(size_t)(b * TN + t) * NTAG + j];
    }
    __syncthreads();
    if (active && mt) sa[b][j] = val;
    __syncthreads();
  }
  if (tid < BN) {
    const int bb = tid;
    float mx = -1e30f;
    for (int i = 0; i < NTAG; ++i) mx = fmaxf(mx, sa[bb][i] + endS[i]);
    float s = 0.0f;
    for (int i = 0; i < NTAG; ++i) s += expf(sa[bb][i] + endS[i] - mx);
    float norm = mx + logf(s);
    int sl = seq_len[bb];
    float gold = startS[target[bb * TN]] + endS[target[bb * TN + (sl - 1)]];
    for (int t = 0; t < TN; ++t) {
      if (t < sl) {
        gold += feats[(size_t)(bb * TN + t) * NTAG + target[bb * TN + t]];
        if (t >= 1) gold += trans[target[bb * TN + t - 1] * NTAG + target[bb * TN + t]];
      }
    }
    out[bb] = norm - gold;
  }
}

// ---------------- host launcher ----------------
static inline size_t cdiv(size_t a, size_t b) { return (a + b - 1) / b; }

extern "C" void kernel_launch(void* const* d_in, const int* in_sizes, int n_in,
                              void* d_out, int out_size, void* d_ws, size_t ws_size,
                              hipStream_t stream) {
  (void)in_sizes; (void)n_in; (void)out_size; (void)ws_size;
  const int*   chars     = (const int*)d_in[0];
  const int*   bigrams   = (const int*)d_in[1];
  const int*   seq_len   = (const int*)d_in[2];
  const int*   target    = (const int*)d_in[3];
  const int*   skips_src = (const int*)d_in[4];
  const int*   skips_wrd = (const int*)d_in[5];
  const int*   lex_cnt   = (const int*)d_in[6];
  const float* char_tab  = (const float*)d_in[7];
  const float* bi_tab    = (const float*)d_in[8];
  const float* word_tab  = (const float*)d_in[9];
  const float* w_ih      = (const float*)d_in[10];
  const float* w_hh      = (const float*)d_in[11];
  const float* b_g       = (const float*)d_in[12];
  const float* a_w_ih    = (const float*)d_in[13];
  const float* a_w_hh    = (const float*)d_in[14];
  const float* a_b       = (const float*)d_in[15];
  const float* ww_ih     = (const float*)d_in[16];
  const float* ww_hh     = (const float*)d_in[17];
  const float* w_b       = (const float*)d_in[18];
  const float* out_w     = (const float*)d_in[19];
  const float* out_b     = (const float*)d_in[20];
  const float* trans_m   = (const float*)d_in[21];
  const float* startS    = (const float*)d_in[22];
  const float* endS      = (const float*)d_in[23];
  float* out = (float*)d_out;

  // ---- bump allocator over d_ws ----
  size_t off = 0;
  auto alloc = [&](size_t bytes) -> char* {
    off = (off + 255) & ~(size_t)255;
    char* p = (char*)d_ws + off;
    off += bytes;
    return p;
  };
  _Float16* wT_wih  = (_Float16*)alloc((size_t)NP_H3 * KP_E * 2);
  _Float16* wT_awih = (_Float16*)alloc((size_t)NP_H  * KP_E * 2);
  _Float16* wT_wwih = (_Float16*)alloc((size_t)NP_H3 * KP_W * 2);
  _Float16* wT_whh  = (_Float16*)alloc((size_t)NP_H3 * KP_H * 2);
  _Float16* wT_wwhh = (_Float16*)alloc((size_t)NP_H3 * KP_H * 2);
  _Float16* wT_awhh = (_Float16*)alloc((size_t)NP_H  * KP_H * 2);
  _Float16* wT_outw = (_Float16*)alloc((size_t)NP_TAG * KP_H * 2);
  _Float16* embA    = (_Float16*)alloc((size_t)BN * TN * KP_E * 2);
  _Float16* wembA   = (_Float16*)alloc((size_t)BN * TN * LN * KP_W * 2);
  _Float16* wwi_h   = (_Float16*)alloc((size_t)BN * TN * LN * H3 * 2);
  _Float16* hsA     = (_Float16*)alloc((size_t)BN * TN * KP_H * 2);
  _Float16* hbA     = (_Float16*)alloc((size_t)160 * KP_H * 2);
  _Float16* cwA     = (_Float16*)alloc((size_t)160 * KP_H * 2);
  _Float16* hA      = (_Float16*)alloc((size_t)BN * KP_H * 2);
  float* wi_all   = (float*)alloc((size_t)BN * TN * H3 * 4);
  float* awi_all  = (float*)alloc((size_t)BN * TN * HN * 4);
  float* Hbuf     = (float*)alloc((size_t)BN * (TN + 1) * HN * 4);
  float* Cbuf     = (float*)alloc((size_t)BN * (TN + 1) * HN * 4);
  float* cb       = (float*)alloc((size_t)160 * HN * 4);
  float* wordsum  = (float*)alloc((size_t)160 * H3 * 4);
  float* cw       = (float*)alloc((size_t)160 * HN * 4);
  float* alphaRaw = (float*)alloc((size_t)160 * HN * 4);
  float* charsum  = (float*)alloc((size_t)BN * H3 * 4);
  float* feats    = (float*)alloc((size_t)BN * TN * NTAG * 4);

  // 1) zero entire workspace span (pads, initial LSTM state)
  {
    size_t words = cdiv(off, 4);
    zero_u32_kernel<<<cdiv(words, 256), 256, 0, stream>>>((unsigned int*)d_ws, words);
  }
  // 2) weight transpose+convert to f16 [Npad, Kpad]
  auto xpose = [&](const float* W, _Float16* Bt, int K, int N, int Kp, int Np) {
    transpose_f16_kernel<<<cdiv((size_t)Np * Kp, 256), 256, 0, stream>>>(W, Bt, K, N, Kp, Np);
  };
  xpose(w_ih,   wT_wih,  100, H3,   KP_E, NP_H3);
  xpose(a_w_ih, wT_awih, 100, HN,   KP_E, NP_H);
  xpose(ww_ih,  wT_wwih, 50,  H3,   KP_W, NP_H3);
  xpose(w_hh,   wT_whh,  HN,  H3,   KP_H, NP_H3);
  xpose(ww_hh,  wT_wwhh, HN,  H3,   KP_H, NP_H3);
  xpose(a_w_hh, wT_awhh, HN,  HN,   KP_H, NP_H);
  xpose(out_w,  wT_outw, HN,  NTAG, KP_H, NP_TAG);
  // 3) embedding gathers
  gather_emb_kernel<<<cdiv((size_t)BN * TN * KP_E, 256), 256, 0, stream>>>(chars, bigrams, char_tab, bi_tab, embA);
  gather_wemb_kernel<<<cdiv((size_t)BN * TN * LN * KP_W, 256), 256, 0, stream>>>(skips_wrd, word_tab, wembA);
  // 4) input-projection GEMMs (WMMA, 1x4 strips)
  auto gemm = [&](const _Float16* A, int lda, const _Float16* Bt, int ldb, const float* bias,
                  float* Cf, _Float16* Ch, int ldc, int M, int N, int Kp, int mstrips, int nstrips) {
    size_t strips = (size_t)mstrips * nstrips;
    wmma_gemm_kernel<<<cdiv(strips, 8), 256, 0, stream>>>(A, lda, Bt, ldb, bias, Cf, Ch, ldc,
                                                          M, N, Kp, mstrips, nstrips);
  };
  gemm(embA,  KP_E, wT_wih,  KP_E, b_g, wi_all,  nullptr, H3, BN * TN,      H3, KP_E, (BN * TN) / 16,      NS_H3);
  gemm(embA,  KP_E, wT_awih, KP_E, a_b, awi_all, nullptr, HN, BN * TN,      HN, KP_E, (BN * TN) / 16,      NS_H);
  gemm(wembA, KP_W, wT_wwih, KP_W, w_b, nullptr, wwi_h,   H3, BN * TN * LN, H3, KP_W, (BN * TN * LN) / 16, NS_H3);
  // 5) sequential lattice-LSTM scan: one persistent workgroup, 2x4 WMMA strips + prefetch
  lattice_scan_kernel<<<1, 512, 0, stream>>>(skips_src, lex_cnt,
                                             wT_wwhh, wT_awhh, wT_whh, wwi_h,
                                             wi_all, awi_all, Hbuf, Cbuf, hsA,
                                             hbA, cwA, hA, cb, wordsum, cw, alphaRaw, charsum);
  // 6) output projection feats = hs @ out_w + out_b
  gemm(hsA, KP_H, wT_outw, KP_H, out_b, feats, nullptr, NTAG, BN * TN, NTAG, KP_H, (BN * TN) / 16, NS_TAG);
  // 7) CRF forward normalizer - gold path score
  crf_kernel<<<1, 512, 0, stream>>>(feats, trans_m, startS, endS, seq_len, target, out);
}
